// EuclideanCodebook_35467839930393
// MI455X (gfx1250) — matrix operand order repaired and checked by reading, not compiled, not run on hardware
//
#include <hip/hip_runtime.h>
#include <hip/hip_bf16.h>

#define DECAY    0.99f
#define EPSILON  1e-05f
#define KCODES   1024
#define DDIM     128
#define NROWS    32768

typedef __attribute__((ext_vector_type(16))) __bf16 v16bf;
typedef __attribute__((ext_vector_type(8)))  float  v8f;

// ---------------------------------------------------------------------------
// Kernel 1: codebook prep — convert embed f32 -> bf16 (once) + ||e||^2.
// ---------------------------------------------------------------------------
__global__ __launch_bounds__(128) void vq_prep_kernel(const float* __restrict__ embed,
                                                      __bf16* __restrict__ ebf,
                                                      float* __restrict__ enorm) {
    __shared__ float red[4];
    const int k = blockIdx.x;
    const int d = threadIdx.x;
    const float v = embed[(size_t)k * DDIM + d];
    ebf[(size_t)k * DDIM + d] = (__bf16)v;
    float s = v * v;
    #pragma unroll
    for (int m = 16; m >= 1; m >>= 1) s += __shfl_xor(s, m, 32);
    if ((d & 31) == 0) red[d >> 5] = s;
    __syncthreads();
    if (d == 0) enorm[k] = red[0] + red[1] + red[2] + red[3];
}

// ---------------------------------------------------------------------------
// B-fragment tile load: 4 chunks x 16 contiguous bf16 per lane (2x b128 each)
// ---------------------------------------------------------------------------
__device__ __forceinline__ void vq_load_btile(v16bf b[4], const __bf16* __restrict__ ebf,
                                              int col, int kb) {
    const __bf16* erow = ebf + (size_t)col * DDIM;
    #pragma unroll
    for (int c = 0; c < 4; ++c) b[c] = *(const v16bf*)(erow + c * 32 + kb);
}

// ---------------------------------------------------------------------------
// Kernel 2: fused distance-GEMM + argmax via WMMA (bf16 A/B, f32 accum).
// One wave = one 16-row tile of x; 64 column tiles of 16 codes, software-
// pipelined with double-buffered B fragments so L2 latency overlaps WMMA.
// dist = 2*(x . e) - ||e||^2   (||x||^2 is row-constant, dropped for argmax)
// ---------------------------------------------------------------------------
__global__ __launch_bounds__(128) void vq_argmax_wmma_kernel(
    const float* __restrict__ x, const __bf16* __restrict__ ebf,
    const float* __restrict__ enorm, int* __restrict__ idx_out,
    float* __restrict__ ind_out) {

    const int wave  = threadIdx.x >> 5;             // 0..3
    const int lane  = threadIdx.x & 31;
    const int row0  = (blockIdx.x * 4 + wave) * 16; // 16-row tile base
    const int r     = lane & 15;                    // row (A) / column (B) in tile
    const int khalf = (lane >> 4) * 16;             // A-fragment K base
    const int kb    = (lane >> 4) * 16;             // B-fragment K base

    // --- Load A fragments once: 4 chunks of K=32 covering D=128 ------------
    v16bf a[4];
    const float* xrow = x + (size_t)(row0 + r) * DDIM;
    #pragma unroll
    for (int c = 0; c < 4; ++c) {
        const float* p0 = xrow + c * 32 + khalf;    // elements 0..7
        const float* p1 = p0 + 8;                   // elements 8..15
        #pragma unroll
        for (int j = 0; j < 8; ++j) {
            a[c][j]     = (__bf16)p0[j];
            a[c][8 + j] = (__bf16)p1[j];
        }
    }

    float best[8];
    int   bidx[8];
    #pragma unroll
    for (int i = 0; i < 8; ++i) { best[i] = -3.402823e38f; bidx[i] = 0; }

    // --- Software-pipelined sweep over 64 column tiles ----------------------
    v16bf b0[4], b1[4];
    float en0, en1;

    vq_load_btile(b0, ebf, r, kb);                  // prologue: tile 0
    en0 = enorm[r];

    for (int ct = 0; ct < KCODES / 16; ct += 2) {
        const int colA = ct * 16 + r;
        const int colB = colA + 16;

        // prefetch tile ct+1 before consuming tile ct
        vq_load_btile(b1, ebf, colB, kb);
        en1 = enorm[colB];

        v8f acc = {};
        #pragma unroll
        for (int c = 0; c < 4; ++c)
            acc = __builtin_amdgcn_wmma_f32_16x16x32_bf16(
                false, a[c], false, b0[c], (short)0, acc, false, false);
        #pragma unroll
        for (int i = 0; i < 8; ++i) {
            float d = fmaf(2.0f, acc[i], -en0);
            if (d > best[i]) { best[i] = d; bidx[i] = colA; }
        }

        // prefetch tile ct+2 (clamped in-bounds) before consuming tile ct+1
        const int nt   = (ct + 2 < KCODES / 16) ? (ct + 2) : ct;
        const int colC = nt * 16 + r;
        vq_load_btile(b0, ebf, colC, kb);
        en0 = enorm[colC];

        v8f acc2 = {};
        #pragma unroll
        for (int c = 0; c < 4; ++c)
            acc2 = __builtin_amdgcn_wmma_f32_16x16x32_bf16(
                false, a[c], false, b1[c], (short)0, acc2, false, false);
        #pragma unroll
        for (int i = 0; i < 8; ++i) {
            float d = fmaf(2.0f, acc2[i], -en1);
            if (d > best[i]) { best[i] = d; bidx[i] = colB; }
        }
    }

    // --- Cross-lane argmax over the 16 lanes of each half (columns) ---------
    #pragma unroll
    for (int m = 1; m < 16; m <<= 1) {
        #pragma unroll
        for (int i = 0; i < 8; ++i) {
            float ov = __shfl_xor(best[i], m, 32);
            int   oi = __shfl_xor(bidx[i], m, 32);
            if (ov > best[i] || (ov == best[i] && oi < bidx[i])) {
                best[i] = ov; bidx[i] = oi;          // first-max tie-break
            }
        }
    }

    // lane 0 holds rows 0..7, lane 16 holds rows 8..15 (C layout)
    if (r == 0) {
        const int base = row0 + (lane >> 4) * 8;
        #pragma unroll
        for (int i = 0; i < 8; ++i) {
            idx_out[base + i] = bidx[i];
            ind_out[base + i] = (float)bidx[i];
        }
    }
}

// ---------------------------------------------------------------------------
// Kernel 3: gather quantize = embed[idx] fused with one-hot scatter sums
// ---------------------------------------------------------------------------
__global__ __launch_bounds__(128) void vq_gather_scatter_kernel(
    const float* __restrict__ x, const float* __restrict__ embed,
    const int* __restrict__ idx, float* __restrict__ quant,
    float* __restrict__ counts, float* __restrict__ esum) {

    const int n = blockIdx.x;
    const int d = threadIdx.x;
    const int k = idx[n];

    const float xv = x[(size_t)n * DDIM + d];
    quant[(size_t)n * DDIM + d] = embed[(size_t)k * DDIM + d];
    atomicAdd(&esum[(size_t)k * DDIM + d], xv);
    if (d == 0) atomicAdd(&counts[k], 1.0f);
}

// ---------------------------------------------------------------------------
// Kernel 4: EMA cluster sizes + total reduction (single block, K=1024)
// ---------------------------------------------------------------------------
__global__ __launch_bounds__(1024) void vq_ema_cs_kernel(
    const float* __restrict__ cs, const float* __restrict__ counts,
    float* __restrict__ ncs, float* __restrict__ total) {

    __shared__ float red[32];
    const int k = threadIdx.x;
    float v = cs[k] * DECAY + (1.0f - DECAY) * counts[k];
    ncs[k] = v;

    #pragma unroll
    for (int m = 16; m >= 1; m >>= 1) v += __shfl_xor(v, m, 32);
    if ((k & 31) == 0) red[k >> 5] = v;
    __syncthreads();
    if (k < 32) {
        float s = red[k];
        #pragma unroll
        for (int m = 16; m >= 1; m >>= 1) s += __shfl_xor(s, m, 32);
        if (k == 0) *total = s;
    }
}

// ---------------------------------------------------------------------------
// Kernel 5: EMA embed_avg + Laplace-smoothed codebook update (K*D elems)
// ---------------------------------------------------------------------------
__global__ __launch_bounds__(256) void vq_embed_update_kernel(
    const float* __restrict__ ea, const float* __restrict__ esum,
    const float* __restrict__ ncs, const float* __restrict__ total,
    float* __restrict__ nea, float* __restrict__ nembed) {

    const int i = blockIdx.x * 256 + threadIdx.x;   // 0 .. K*D-1
    const int k = i >> 7;                           // /128
    const float t = *total;

    const float v = ea[i] * DECAY + (1.0f - DECAY) * esum[i];
    nea[i] = v;
    const float smoothed = (ncs[k] + EPSILON) / (t + KCODES * EPSILON) * t;
    nembed[i] = v / smoothed;
}

// ---------------------------------------------------------------------------
extern "C" void kernel_launch(void* const* d_in, const int* in_sizes, int n_in,
                              void* d_out, int out_size, void* d_ws, size_t ws_size,
                              hipStream_t stream) {
    const float* x     = (const float*)d_in[0];  // (8,4096,128)
    const float* embed = (const float*)d_in[1];  // (1024,128)
    const float* cs    = (const float*)d_in[2];  // (1024,)
    const float* ea    = (const float*)d_in[3];  // (1024,128)

    // Outputs concatenated flat in return order (all as f32 buffer)
    float* out    = (float*)d_out;
    float* quant  = out;                                   // N*D = 4194304
    float* ind    = quant + (size_t)NROWS * DDIM;          // N   = 32768
    float* ncs    = ind + NROWS;                           // K   = 1024
    float* nea    = ncs + KCODES;                          // K*D = 131072
    float* nembed = nea + (size_t)KCODES * DDIM;           // K*D = 131072

    // Workspace layout
    char*   ws     = (char*)d_ws;
    int*    idx    = (int*)ws;                             // N ints
    float*  counts = (float*)(ws + (size_t)NROWS * 4);     // K
    float*  esum   = counts + KCODES;                      // K*D
    float*  enorm  = esum + (size_t)KCODES * DDIM;         // K
    float*  total  = enorm + KCODES;                       // 1 (+pad)
    __bf16* ebf    = (__bf16*)(total + 16);                // K*D bf16 (256 KB)

    // Zero the scatter accumulators every call (deterministic / graph-safe)
    hipMemsetAsync(counts, 0, (size_t)(KCODES + KCODES * DDIM) * sizeof(float), stream);

    vq_prep_kernel<<<KCODES, 128, 0, stream>>>(embed, ebf, enorm);
    vq_argmax_wmma_kernel<<<NROWS / 64, 128, 0, stream>>>(x, ebf, enorm, idx, ind);
    vq_gather_scatter_kernel<<<NROWS, 128, 0, stream>>>(x, embed, idx, quant, counts, esum);
    vq_ema_cs_kernel<<<1, 1024, 0, stream>>>(cs, counts, ncs, total);
    vq_embed_update_kernel<<<(KCODES * DDIM) / 256, 256, 0, stream>>>(
        ea, esum, ncs, total, nea, nembed);
}
